// QuantizedLinear_83786222010511
// MI455X (gfx1250) — matrix operand order repaired and checked by reading, not compiled, not run on hardware
//
#include <hip/hip_runtime.h>

typedef _Float16 v16h __attribute__((ext_vector_type(16)));
typedef _Float16 h2   __attribute__((ext_vector_type(2)));
typedef float    v8f  __attribute__((ext_vector_type(8)));
typedef int      v4i  __attribute__((ext_vector_type(4)));
typedef int      v8i  __attribute__((ext_vector_type(8)));

#define IN_F   4096
#define OUT_F  16384
#define BATCH  32

// ---------------------------------------------------------------------------
// Kernel 1: convert x fp32 -> f16 into workspace (one pass, 256 KB total).
// ---------------------------------------------------------------------------
__global__ __launch_bounds__(256)
void QuantizedLinear_cvt_x(const float* __restrict__ x, _Float16* __restrict__ xh) {
    int i = (blockIdx.x * blockDim.x + threadIdx.x) * 8;
    float4 v0 = *(const float4*)(x + i);
    float4 v1 = *(const float4*)(x + i + 4);
    h2 p0 = { (_Float16)v0.x, (_Float16)v0.y };
    h2 p1 = { (_Float16)v0.z, (_Float16)v0.w };
    h2 p2 = { (_Float16)v1.x, (_Float16)v1.y };
    h2 p3 = { (_Float16)v1.z, (_Float16)v1.w };
    v4i o;
    o[0] = __builtin_bit_cast(int, p0);
    o[1] = __builtin_bit_cast(int, p1);
    o[2] = __builtin_bit_cast(int, p2);
    o[3] = __builtin_bit_cast(int, p3);
    *(v4i*)(xh + i) = o;
}

// ---------------------------------------------------------------------------
// int8x16 -> f16x16, branch/cvt-free:
//   t = b ^ 0x80  (=> b+128 in [0,255])
//   f16 bits (0x6400 | t) == 1024 + t  (exact: integer range of f16 @ exp 2^10)
//   subtract 1152 (packed)             => exact signed value
// Per dword: 1 xor + 2 v_perm_b32 + 2 v_pk_add_f16 = 5 VALU for 4 elements.
// Output dwords are already in WMMA B-operand half ordering.
// ---------------------------------------------------------------------------
__device__ __forceinline__ v16h cvt_i8x16_f16(v4i wb) {
    const h2 kBias = { (_Float16)-1152.0f, (_Float16)-1152.0f };
    v8i bi;
#pragma unroll
    for (int d = 0; d < 4; d++) {
        unsigned t  = ((unsigned)wb[d]) ^ 0x80808080u;
        unsigned lo = __builtin_amdgcn_perm(0x64646464u, t, 0x04010400u); // {64,b1,64,b0}
        unsigned hi = __builtin_amdgcn_perm(0x64646464u, t, 0x04030402u); // {64,b3,64,b2}
        h2 flo = __builtin_bit_cast(h2, lo) + kBias;
        h2 fhi = __builtin_bit_cast(h2, hi) + kBias;
        bi[2*d]     = __builtin_bit_cast(int, flo);
        bi[2*d + 1] = __builtin_bit_cast(int, fhi);
    }
    return __builtin_bit_cast(v16h, bi);
}

// ---------------------------------------------------------------------------
// Kernel 2: y = x_f16 @ W_int8^T * scale, via v_wmma_f32_16x16x32_f16.
// Wave tile: M=32 (both batch tiles) x N=16; block = 4 waves -> 32x64.
// One B conversion feeds two WMMAs. All per-lane loads are contiguous 16B.
// ---------------------------------------------------------------------------
__global__ __launch_bounds__(128)
void QuantizedLinear_wmma(const _Float16* __restrict__ xh,
                          const signed char* __restrict__ W,
                          const float* __restrict__ scale,
                          float* __restrict__ out) {
    const int lane  = threadIdx.x & 31;
    const int wave  = threadIdx.x >> 5;
    const int col   = lane & 15;     // N index within 16-wide tile
    const int hi    = lane >> 4;     // half-wave select
    const int o     = blockIdx.x * 64 + wave * 16 + col;   // output channel

    // A-matrix per-lane base pointers (f16, 16B-aligned)
    const _Float16* aRow0 = xh + (size_t)col        * IN_F + hi * 16;  // M 0..15
    const _Float16* aRow1 = xh + (size_t)(col + 16) * IN_F + hi * 16;  // M 16..31
    // B-matrix per-lane base pointer (int8, 16B-aligned)
    const signed char* wRow = W + (size_t)o * IN_F + hi * 16;

    v8f c0 = {}; v8f c1 = {};

#pragma unroll 4
    for (int k = 0; k < IN_F; k += 32) {
        // ---- A tiles: 4 x global_load_b128 (dword-granular assembly) ----
        v4i a0l = *(const v4i*)(aRow0 + k);
        v4i a0h = *(const v4i*)(aRow0 + k + 8);
        v4i a1l = *(const v4i*)(aRow1 + k);
        v4i a1h = *(const v4i*)(aRow1 + k + 8);

        // ---- B tile: 1 x non-temporal global_load_b128 of raw int8 ----
        v4i wb = __builtin_nontemporal_load((const v4i*)(wRow + k));

        v8i a0i, a1i;
#pragma unroll
        for (int i = 0; i < 4; i++) {
            a0i[i] = a0l[i]; a0i[i + 4] = a0h[i];
            a1i[i] = a1l[i]; a1i[i + 4] = a1h[i];
        }
        v16h A0 = __builtin_bit_cast(v16h, a0i);
        v16h A1 = __builtin_bit_cast(v16h, a1i);
        v16h B  = cvt_i8x16_f16(wb);

        // D = A x B + C (f32 accumulate), B shared across both M tiles
        c0 = __builtin_amdgcn_wmma_f32_16x16x32_f16(false, A0, false, B,
                                                    (short)0, c0, false, false);
        c1 = __builtin_amdgcn_wmma_f32_16x16x32_f16(false, A1, false, B,
                                                    (short)0, c1, false, false);
    }

    // ---- Epilogue: per-output-channel scale, fp32 store ----
    const float s = scale[o];
#pragma unroll
    for (int r = 0; r < 8; r++) {
        const int m = r + hi * 8;
        out[(size_t)m        * OUT_F + o] = c0[r] * s;   // batch rows 0..15
        out[(size_t)(m + 16) * OUT_F + o] = c1[r] * s;   // batch rows 16..31
    }
}

// ---------------------------------------------------------------------------
extern "C" void kernel_launch(void* const* d_in, const int* in_sizes, int n_in,
                              void* d_out, int out_size, void* d_ws, size_t ws_size,
                              hipStream_t stream) {
    (void)in_sizes; (void)n_in; (void)out_size; (void)ws_size;

    const float*       x     = (const float*)d_in[0];
    const signed char* W     = (const signed char*)d_in[1];
    const float*       scale = (const float*)d_in[2];
    float*             out   = (float*)d_out;
    _Float16*          xh    = (_Float16*)d_ws;          // 32*4096*2B = 256 KB

    QuantizedLinear_cvt_x<<<(BATCH * IN_F) / (256 * 8), 256, 0, stream>>>(x, xh);
    QuantizedLinear_wmma<<<OUT_F / 64, 128, 0, stream>>>(xh, W, scale, out);
}
